// DiffusionTransformer_14104672600296
// MI455X (gfx1250) — compile-verified, matrix operand order checked by reading, not run on hardware
//
#include <hip/hip_runtime.h>
#include <hip/hip_bf16.h>

typedef __bf16 bf16_t;
typedef __attribute__((ext_vector_type(16))) __bf16 v16bf;
typedef __attribute__((ext_vector_type(8)))  __bf16 v8bf;
typedef __attribute__((ext_vector_type(8)))  float   v8f;

#define Bc   2
#define Nc   1024
#define Cc   384
#define CPc  16
#define Hc   16
#define Dc   24
#define NBc  24
#define HIDc 768
#define Mc   (Bc*Nc)
#define EPSc 1e-5f

// ---- CDNA5 16x16x32 bf16 WMMA fragment K map (ISA 7.12.2) ----
// lane holds row/col = lane&15; half = lane>>4.
// element e: e<8 -> K = 8*half + e (contiguous); e>=8 -> K = 16 + 8*half + (e-8).
__device__ __forceinline__ int kmap(int e, int half) {
  int r = e >> 1;
  int base = (r < 4) ? (2 * r) : (16 + 2 * (r - 4));
  return base + 8 * half + (e & 1);
}
__device__ __forceinline__ float sigmoidf_(float x) { return 1.f / (1.f + __expf(-x)); }
__device__ __forceinline__ v8f wmma_bf16(v16bf a, v16bf b, v8f c) {
  return __builtin_amdgcn_wmma_f32_16x16x32_bf16(false, a, false, b, (short)0, c,
                                                 false, false);
}
__device__ __forceinline__ v16bf cat8(v8bf lo, v8bf hi) {
  return __builtin_shufflevector(lo, hi, 0, 1, 2, 3, 4, 5, 6, 7,
                                 8, 9, 10, 11, 12, 13, 14, 15);
}
// A fragment: two contiguous 16B loads (K-map contiguity per half-lane)
__device__ __forceinline__ v16bf load_a_frag(const bf16_t* __restrict__ A,
                                             size_t rowbase, int k0, int half) {
  v8bf lo = *(const v8bf*)(A + rowbase + k0 + 8 * half);
  v8bf hi = *(const v8bf*)(A + rowbase + k0 + 16 + 8 * half);
  return cat8(lo, hi);
}
// B fragment from pre-packed weights: 32 contiguous bytes per lane per tile
__device__ __forceinline__ v16bf load_b_frag(const bf16_t* __restrict__ Wp,
                                             size_t tbase, int lane) {
  const bf16_t* p = Wp + tbase + (size_t)lane * 16;
  v8bf lo = *(const v8bf*)p;
  v8bf hi = *(const v8bf*)(p + 8);
  return cat8(lo, hi);
}

// ---------------- LayerNorm over last dim C ----------------
__global__ __launch_bounds__(128) void ln_k(const float* __restrict__ x,
                                            float* __restrict__ y) {
  __shared__ float r1[128];
  __shared__ float r2[128];
  int row = blockIdx.x;
  const float* xr = x + (size_t)row * Cc;
  float s = 0.f, s2 = 0.f;
  for (int c = threadIdx.x; c < Cc; c += 128) { float v = xr[c]; s += v; s2 += v * v; }
  r1[threadIdx.x] = s; r2[threadIdx.x] = s2;
  __syncthreads();
  for (int off = 64; off > 0; off >>= 1) {
    if (threadIdx.x < off) { r1[threadIdx.x] += r1[threadIdx.x + off];
                             r2[threadIdx.x] += r2[threadIdx.x + off]; }
    __syncthreads();
  }
  float mean = r1[0] / (float)Cc;
  float var  = r2[0] / (float)Cc - mean * mean;
  float inv  = rsqrtf(var + EPSc);
  for (int c = threadIdx.x; c < Cc; c += 128)
    y[(size_t)row * Cc + c] = (xr[c] - mean) * inv;
}

// ---------------- elementwise helpers ----------------
__global__ __launch_bounds__(256) void scale_bf16_k(const float* __restrict__ x,
                                                    const float* __restrict__ w,
                                                    bf16_t* __restrict__ y) {
  int i = blockIdx.x * 256 + threadIdx.x;
  if (i < Mc * Cc) y[i] = (bf16_t)(x[i] * w[i % Cc]);
}
__global__ __launch_bounds__(256) void mul_bf16_k(const float* __restrict__ a,
                                                  const float* __restrict__ b,
                                                  bf16_t* __restrict__ y) {
  int i = blockIdx.x * 256 + threadIdx.x;
  if (i < Mc * Cc) y[i] = (bf16_t)(a[i] * b[i]);
}
__global__ __launch_bounds__(256) void copy_k(const float* __restrict__ x,
                                              float* __restrict__ y, int n) {
  int i = blockIdx.x * 256 + threadIdx.x;
  if (i < n) y[i] = x[i];
}

// ------- pack f32 weights [NB,K,N] -> bf16 WMMA-fragment order -------
// Wp[((kt*tilesN + nt)*32 + lane)*16 + e] = W[kt*32 + kmap(e,half)][nt*16 + lx]
__global__ __launch_bounds__(256) void pack_w_k(const float* __restrict__ W,
                                                bf16_t* __restrict__ Wp,
                                                int K, int N, long total) {
  long g = (long)blockIdx.x * 256 + threadIdx.x;
  if (g >= total) return;
  long per = (long)K * N;
  int blk = (int)(g / per);
  long p = g - (long)blk * per;
  int e = (int)(p & 15);
  int lane = (int)((p >> 4) & 31);
  long tile = p >> 9;
  int tilesN = N >> 4;
  int kt = (int)(tile / tilesN), nt = (int)(tile % tilesN);
  int lx = lane & 15, half = lane >> 4;
  int kk = kt * 32 + kmap(e, half);
  int n  = nt * 16 + lx;
  Wp[g] = (bf16_t)W[(size_t)blk * per + (size_t)kk * N + n];
}

// ------- generic WMMA GEMM: 16x64 per wave, async A staging, packed weights -------
// OMODE: 0 = f32 row-major, 1 = bf16 row-major, 2 = bf16 V-transposed [b,h,d,n]
// Requires N % 64 == 0.
template <bool HAS_BIAS, bool SIG, bool HAS_AUX, bool ACCUM, int OMODE>
__global__ __launch_bounds__(256) void gemm_k(const bf16_t* __restrict__ A,
                                              const bf16_t* __restrict__ Wp,
                                              const float* __restrict__ bias,
                                              const float* __restrict__ aux,
                                              float* __restrict__ outF,
                                              bf16_t* __restrict__ outB,
                                              int M, int K, int N) {
  __shared__ __align__(16) bf16_t astage[8][2][16 * 32];  // double-buffered A tiles
  int lane = threadIdx.x & 31, wave = threadIdx.x >> 5;
  int tile = blockIdx.x * 8 + wave;
  int tilesN = N >> 4;
  int colTiles = N >> 6;                       // 4 n-tiles per wave
  int tiles = (M >> 4) * colTiles;
  if (tile >= tiles) return;
  int m0 = (tile / colTiles) << 4;
  int ng = (tile % colTiles) << 2;             // first 16-wide n tile index
  int lx = lane & 15, half = lane >> 4;
  int srow = lane >> 2, sseg = lane & 3;
  v8f acc[4] = {{}, {}, {}, {}};
  for (int k0 = 0; k0 < K; k0 += 32) {
    int buf = (k0 >> 5) & 1;
    unsigned lbase = (unsigned)(size_t)&astage[wave][buf][0];
    // stage 16x32 bf16 A tile (1 KB) into LDS: 2 async b128 per lane
#pragma unroll
    for (int c = 0; c < 2; ++c) {
      int rrow = srow + c * 8;
      unsigned long long ga =
          (unsigned long long)(size_t)(A + (size_t)(m0 + rrow) * K + k0 + sseg * 8);
      unsigned la = lbase + (unsigned)(rrow * 64 + sseg * 16);
      asm volatile("global_load_async_to_lds_b128 %0, %1, off"
                   :: "v"(la), "v"(ga) : "memory");
    }
    size_t ktbase = (size_t)(k0 >> 5) * tilesN;
    if (k0 + 32 < K)
      __builtin_prefetch(Wp + ((ktbase + tilesN + ng) << 9), 0, 0);
    asm volatile("s_wait_asynccnt 0x0" ::: "memory");
    v8bf alo = *(const v8bf*)(&astage[wave][buf][lx * 32 + 8 * half]);
    v8bf ahi = *(const v8bf*)(&astage[wave][buf][lx * 32 + 16 + 8 * half]);
    v16bf a = cat8(alo, ahi);
#pragma unroll
    for (int t = 0; t < 4; ++t) {
      v16bf b = load_b_frag(Wp, (ktbase + ng + t) << 9, lane);
      acc[t] = wmma_bf16(a, b, acc[t]);
    }
  }
#pragma unroll
  for (int t = 0; t < 4; ++t) {
    int col = ((ng + t) << 4) + lx;
    float bv = HAS_BIAS ? bias[col] : 0.f;
#pragma unroll
    for (int r = 0; r < 8; ++r) {
      int m = m0 + r + 8 * half;
      float v = acc[t][r] + bv;
      if (SIG) v = sigmoidf_(v);
      if (HAS_AUX) v *= aux[(size_t)m * N + col];
      if constexpr (OMODE == 0) {
        size_t idx = (size_t)m * N + col;
        if (ACCUM) outF[idx] += v; else outF[idx] = v;
      } else if constexpr (OMODE == 1) {
        outB[(size_t)m * N + col] = (bf16_t)v;
      } else {  // V transposed: [b,h,d,token]
        int bb = m / Nc, tok = m % Nc;
        int hh = col / Dc, dd = col % Dc;
        outB[(((size_t)bb * Hc + hh) * Dc + dd) * Nc + tok] = (bf16_t)v;
      }
    }
  }
}

// ---------- fused AdaLN: out = sigmoid(A@W1 + b1) * aux + A@W2 (bf16) ----------
__global__ __launch_bounds__(256) void adaln_gemm_k(const bf16_t* __restrict__ A,
                                                    const bf16_t* __restrict__ W1p,
                                                    const float* __restrict__ b1,
                                                    const bf16_t* __restrict__ W2p,
                                                    const float* __restrict__ aux,
                                                    bf16_t* __restrict__ out,
                                                    int M, int K, int N) {
  int lane = threadIdx.x & 31, wave = threadIdx.x >> 5;
  int tile = blockIdx.x * 8 + wave;
  int tilesN = N >> 4, tiles = (M >> 4) * tilesN;
  if (tile >= tiles) return;
  int m0 = (tile / tilesN) << 4, n0 = (tile % tilesN) << 4;
  int lx = lane & 15, half = lane >> 4;
  size_t rowbase = (size_t)(m0 + lx) * K;
  int col = n0 + lx;
  v8f acc1 = {}, acc2 = {};
  for (int k0 = 0; k0 < K; k0 += 32) {
    size_t tbase = ((((size_t)(k0 >> 5)) * tilesN + (size_t)(n0 >> 4)) << 9);
    v16bf a = load_a_frag(A, rowbase, k0, half);
    acc1 = wmma_bf16(a, load_b_frag(W1p, tbase, lane), acc1);
    acc2 = wmma_bf16(a, load_b_frag(W2p, tbase, lane), acc2);
  }
  float bv = b1[col];
#pragma unroll
  for (int r = 0; r < 8; ++r) {
    int m = m0 + r + 8 * half;
    float v = sigmoidf_(acc1[r] + bv) * aux[(size_t)m * N + col] + acc2[r];
    out[(size_t)m * N + col] = (bf16_t)v;
  }
}

// ---------- fused SwiGLU: out = silu(A@W1) * (A@W2) (bf16, N=HID) ----------
__global__ __launch_bounds__(256) void swiglu_gemm_k(const bf16_t* __restrict__ A,
                                                     const bf16_t* __restrict__ W1p,
                                                     const bf16_t* __restrict__ W2p,
                                                     bf16_t* __restrict__ out,
                                                     int M, int K, int N) {
  int lane = threadIdx.x & 31, wave = threadIdx.x >> 5;
  int tile = blockIdx.x * 8 + wave;
  int tilesN = N >> 4, tiles = (M >> 4) * tilesN;
  if (tile >= tiles) return;
  int m0 = (tile / tilesN) << 4, n0 = (tile % tilesN) << 4;
  int lx = lane & 15, half = lane >> 4;
  size_t rowbase = (size_t)(m0 + lx) * K;
  int col = n0 + lx;
  v8f acc1 = {}, acc2 = {};
  for (int k0 = 0; k0 < K; k0 += 32) {
    size_t tbase = ((((size_t)(k0 >> 5)) * tilesN + (size_t)(n0 >> 4)) << 9);
    v16bf a = load_a_frag(A, rowbase, k0, half);
    acc1 = wmma_bf16(a, load_b_frag(W1p, tbase, lane), acc1);
    acc2 = wmma_bf16(a, load_b_frag(W2p, tbase, lane), acc2);
  }
#pragma unroll
  for (int r = 0; r < 8; ++r) {
    int m = m0 + r + 8 * half;
    float x = acc1[r];
    out[(size_t)m * N + col] = (bf16_t)(x * sigmoidf_(x) * acc2[r]);
  }
}

// ---------- pair bias: zb[b,h,i,j] = LN16(z[b,i,j,:]) @ (diag(plw)@pbw) + plb@pbw ----------
__global__ __launch_bounds__(256) void zbias_k(const float* __restrict__ z,
                                               const float* __restrict__ plw,
                                               const float* __restrict__ plb,
                                               const float* __restrict__ pbw,
                                               bf16_t* __restrict__ zb) {
  int lane = threadIdx.x & 31, wave = threadIdx.x >> 5;
  long w = (long)blockIdx.x * 8 + wave;
  long p0 = w * 16;
  int lx = lane & 15, half = lane >> 4;
  long p = p0 + lx;
  int b0  = (int)(p0 / ((long)Nc * Nc));
  long r0 = p0 - (long)b0 * Nc * Nc;
  int i0  = (int)(r0 / Nc);
  int j0  = (int)(r0 % Nc);
  const float* zr = z + p * CPc;
  float v[CPc];
  float s = 0.f;
#pragma unroll
  for (int c = 0; c < CPc; ++c) { v[c] = zr[c]; s += v[c]; }
  float mean = s * (1.f / CPc), s2 = 0.f;
#pragma unroll
  for (int c = 0; c < CPc; ++c) { float d = v[c] - mean; v[c] = d; s2 += d * d; }
  float inv = rsqrtf(s2 * (1.f / CPc) + EPSc);
  v16bf a, bb;
#pragma unroll
  for (int e = 0; e < 16; ++e) {
    int kk = kmap(e, half);
    a[e]  = (kk < CPc) ? (bf16_t)(v[kk] * inv) : (bf16_t)0.f;
    bb[e] = (kk < CPc) ? (bf16_t)(plw[kk] * pbw[kk * Hc + lx]) : (bf16_t)0.f;
  }
  v8f acc = {};
  acc = wmma_bf16(a, bb, acc);
  float bh = 0.f;
#pragma unroll
  for (int c = 0; c < CPc; ++c) bh += plb[c] * pbw[c * Hc + lx];
#pragma unroll
  for (int r = 0; r < 8; ++r) {
    int m = r + 8 * half;
    long idx = (((long)b0 * Hc + lx) * Nc + i0) * Nc + (j0 + m);
    zb[idx] = (bf16_t)(acc[r] + bh);
  }
}

// ---------- attention: one wave per (b,h,16-query tile); vt = V^T [b,h,d,n] ----------
__global__ __launch_bounds__(32) void attn_k(const bf16_t* __restrict__ q,
                                             const bf16_t* __restrict__ k,
                                             const bf16_t* __restrict__ vt,
                                             const bf16_t* __restrict__ zb,
                                             const float* __restrict__ mask,
                                             float* __restrict__ o) {
  __shared__ __align__(16) bf16_t lsc[16 * Nc];   // 32 KB score strip
  __shared__ float rmax[16];
  __shared__ float rinv[16];
  int lane = threadIdx.x;
  int lx = lane & 15, half = lane >> 4;
  int wg = blockIdx.x;
  int qt = wg % (Nc / 16);
  int bh = wg / (Nc / 16);
  int h = bh % Hc, b = bh / Hc;
  int q0 = qt * 16;
  const float inv_sqrt_d = rsqrtf((float)Dc);
  const bf16_t* qrow = q + (size_t)(b * Nc + q0 + lx) * Cc + h * Dc;
  v8bf qlo = *(const v8bf*)(qrow + 8 * half);
  v8bf qhi = {};
  if (half == 0) qhi = *(const v8bf*)(qrow + 16);
  v16bf aq = cat8(qlo, qhi);
  for (int kt = 0; kt < Nc / 16; ++kt) {
    const bf16_t* krow = k + (size_t)(b * Nc + kt * 16 + lx) * Cc + h * Dc;
    v8bf klo = *(const v8bf*)(krow + 8 * half);
    v8bf khi = {};
    if (half == 0) khi = *(const v8bf*)(krow + 16);
    v8f acc = {};
    acc = wmma_bf16(aq, cat8(klo, khi), acc);
    float mb = (mask[b * Nc + kt * 16 + lx] - 1.f) * 1e9f;
#pragma unroll
    for (int r = 0; r < 8; ++r) {
      int m = r + 8 * half;
      float zv = (float)zb[(((long)b * Hc + h) * Nc + (q0 + m)) * Nc + kt * 16 + lx];
      lsc[m * Nc + kt * 16 + lx] = (bf16_t)(acc[r] * inv_sqrt_d + zv + mb);
    }
  }
  __syncthreads();
  if (lane < 16) {
    float mx = -1e30f;
    for (int j = 0; j < Nc; ++j) mx = fmaxf(mx, (float)lsc[lane * Nc + j]);
    float sm = 0.f;
    for (int j = 0; j < Nc; ++j) sm += __expf((float)lsc[lane * Nc + j] - mx);
    rmax[lane] = mx; rinv[lane] = 1.f / sm;
  }
  __syncthreads();
  for (int idx = lane; idx < 16 * Nc; idx += 32) {
    int rr = idx >> 10;
    lsc[idx] = (bf16_t)(__expf((float)lsc[idx] - rmax[rr]) * rinv[rr]);
  }
  __syncthreads();
  v8f o0 = {}, o1 = {};
  size_t vbase = ((size_t)(b * Hc + h)) * Dc * Nc;
  for (int kt = 0; kt < Nc / 32; ++kt) {
    int kcol = kt * 32 + 8 * half;
    v8bf plo = *(const v8bf*)(&lsc[lx * Nc + kcol]);
    v8bf phi = *(const v8bf*)(&lsc[lx * Nc + kcol + 16]);
    v16bf ap = cat8(plo, phi);
    const bf16_t* v0r = vt + vbase + (size_t)lx * Nc + kcol;
    v16bf bv0 = cat8(*(const v8bf*)v0r, *(const v8bf*)(v0r + 16));
    v16bf bv1;
    if (lx < 8) {
      const bf16_t* v1r = vt + vbase + (size_t)(16 + lx) * Nc + kcol;
      bv1 = cat8(*(const v8bf*)v1r, *(const v8bf*)(v1r + 16));
    } else {
      v8bf zz = {};
      bv1 = cat8(zz, zz);
    }
    o0 = wmma_bf16(ap, bv0, o0);
    o1 = wmma_bf16(ap, bv1, o1);
  }
#pragma unroll
  for (int r = 0; r < 8; ++r) {
    int m = r + 8 * half;
    size_t orow = (size_t)(b * Nc + q0 + m) * Cc + h * Dc;
    o[orow + lx] = o0[r];
    if (16 + lx < Dc) o[orow + 16 + lx] = o1[r];
  }
}

// ====================== host side ======================
extern "C" void kernel_launch(void* const* d_in, const int* in_sizes, int n_in,
                              void* d_out, int out_size, void* d_ws, size_t ws_size,
                              hipStream_t stream) {
  (void)in_sizes; (void)n_in; (void)ws_size;
  const float* single_repr = (const float*)d_in[0];
  const float* single_proj = (const float*)d_in[1];
  const float* pair_repr   = (const float*)d_in[2];
  const float* mask        = (const float*)d_in[3];
  const float* adaln_s_w   = (const float*)d_in[4];
  const float* gate_w      = (const float*)d_in[5];
  const float* gate_b      = (const float*)d_in[6];
  const float* skip_w      = (const float*)d_in[7];
  const float* q_w         = (const float*)d_in[8];
  const float* q_b         = (const float*)d_in[9];
  const float* k_w         = (const float*)d_in[10];
  const float* v_w         = (const float*)d_in[11];
  const float* pair_ln_w   = (const float*)d_in[12];
  const float* pair_ln_b   = (const float*)d_in[13];
  const float* pair_b_w    = (const float*)d_in[14];
  const float* gating_w    = (const float*)d_in[15];
  const float* out_w       = (const float*)d_in[16];
  const float* s_gate_w    = (const float*)d_in[17];
  const float* s_gate_b    = (const float*)d_in[18];
  const float* t_adaln_s_w = (const float*)d_in[19];
  const float* t_gate_w    = (const float*)d_in[20];
  const float* t_gate_b    = (const float*)d_in[21];
  const float* t_skip_w    = (const float*)d_in[22];
  const float* w1a         = (const float*)d_in[23];
  const float* w1b         = (const float*)d_in[24];
  const float* w2          = (const float*)d_in[25];
  const float* t_s_gate_w  = (const float*)d_in[26];
  const float* t_s_gate_b  = (const float*)d_in[27];

  char* ws = (char*)d_ws;
  size_t off = 0;
  auto take = [&](size_t bytes) -> char* {
    char* p = ws + off;
    off = (off + bytes + 255) & ~(size_t)255;
    return p;
  };
  const size_t fMC = (size_t)Mc * Cc * sizeof(float);
  const size_t bMC = (size_t)Mc * Cc * sizeof(bf16_t);
  const size_t bMH = (size_t)Mc * HIDc * sizeof(bf16_t);
  const size_t bVT = (size_t)Bc * Hc * Dc * Nc * sizeof(bf16_t);
  const size_t bZB = (size_t)Bc * Hc * Nc * Nc * sizeof(bf16_t);
  const size_t pCC = (size_t)NBc * Cc * Cc * sizeof(bf16_t);
  const size_t pCH = (size_t)NBc * Cc * HIDc * sizeof(bf16_t);
  const size_t pHC = (size_t)NBc * HIDc * Cc * sizeof(bf16_t);

  float*  a_buf  = (float*)take(fMC);
  float*  a_ln   = (float*)take(fMC);
  float*  s_ln   = (float*)take(fMC);
  float*  gbuf   = (float*)take(fMC);
  float*  sg     = (float*)take(fMC);
  float*  tsg    = (float*)take(fMC);
  float*  obuf   = (float*)take(fMC);
  bf16_t* s_n    = (bf16_t*)take(bMC);
  bf16_t* s_t    = (bf16_t*)take(bMC);
  bf16_t* a_n    = (bf16_t*)take(bMC);
  bf16_t* a_t    = (bf16_t*)take(bMC);
  bf16_t* qb     = (bf16_t*)take(bMC);
  bf16_t* kb     = (bf16_t*)take(bMC);
  bf16_t* vtb    = (bf16_t*)take(bVT);
  bf16_t* ogb    = (bf16_t*)take(bMC);
  bf16_t* hidden = (bf16_t*)take(bMH);
  bf16_t* P_gw   = (bf16_t*)take(pCC);
  bf16_t* P_sk   = (bf16_t*)take(pCC);
  bf16_t* P_qw   = (bf16_t*)take(pCC);
  bf16_t* P_kw   = (bf16_t*)take(pCC);
  bf16_t* P_vw   = (bf16_t*)take(pCC);
  bf16_t* P_gat  = (bf16_t*)take(pCC);
  bf16_t* P_ow   = (bf16_t*)take(pCC);
  bf16_t* P_sgw  = (bf16_t*)take(pCC);
  bf16_t* P_tgw  = (bf16_t*)take(pCC);
  bf16_t* P_tsk  = (bf16_t*)take(pCC);
  bf16_t* P_tsgw = (bf16_t*)take(pCC);
  bf16_t* P_w1a  = (bf16_t*)take(pCH);
  bf16_t* P_w1b  = (bf16_t*)take(pCH);
  bf16_t* P_w2   = (bf16_t*)take(pHC);
  bf16_t* zbuf   = (bf16_t*)take(bZB);

  const int ELT_B    = (Mc * Cc + 255) / 256;
  const int GEMM64_C = (Mc / 16) * (Cc / 64) / 8;   // 96 (16x64-tile gemm_k, N=384)
  const int ADA_CC   = (Mc / 16) * (Cc / 16) / 8;   // 384
  const int SWI_CH   = (Mc / 16) * (HIDc / 16) / 8; // 768
  const int ZB_B     = (Bc * Nc * Nc / 16) / 8;     // 16384
  const int ATTN_B   = Bc * Hc * (Nc / 16);         // 2048
  const long totCC = (long)NBc * Cc * Cc;
  const long totCH = (long)NBc * Cc * HIDc;
  const long totHC = (long)NBc * HIDc * Cc;
  const int PB_CC = (int)((totCC + 255) / 256);
  const int PB_CH = (int)((totCH + 255) / 256);
  const int PB_HC = (int)((totHC + 255) / 256);

  // one-time: shared LN + running buffer + weight packing
  ln_k<<<Mc, 128, 0, stream>>>(single_proj, s_ln);
  copy_k<<<ELT_B, 256, 0, stream>>>(single_repr, a_buf, Mc * Cc);
  pack_w_k<<<PB_CC, 256, 0, stream>>>(gate_w,     P_gw,   Cc, Cc,  totCC);
  pack_w_k<<<PB_CC, 256, 0, stream>>>(skip_w,     P_sk,   Cc, Cc,  totCC);
  pack_w_k<<<PB_CC, 256, 0, stream>>>(q_w,        P_qw,   Cc, Cc,  totCC);
  pack_w_k<<<PB_CC, 256, 0, stream>>>(k_w,        P_kw,   Cc, Cc,  totCC);
  pack_w_k<<<PB_CC, 256, 0, stream>>>(v_w,        P_vw,   Cc, Cc,  totCC);
  pack_w_k<<<PB_CC, 256, 0, stream>>>(gating_w,   P_gat,  Cc, Cc,  totCC);
  pack_w_k<<<PB_CC, 256, 0, stream>>>(out_w,      P_ow,   Cc, Cc,  totCC);
  pack_w_k<<<PB_CC, 256, 0, stream>>>(s_gate_w,   P_sgw,  Cc, Cc,  totCC);
  pack_w_k<<<PB_CC, 256, 0, stream>>>(t_gate_w,   P_tgw,  Cc, Cc,  totCC);
  pack_w_k<<<PB_CC, 256, 0, stream>>>(t_skip_w,   P_tsk,  Cc, Cc,  totCC);
  pack_w_k<<<PB_CC, 256, 0, stream>>>(t_s_gate_w, P_tsgw, Cc, Cc,  totCC);
  pack_w_k<<<PB_CH, 256, 0, stream>>>(w1a,        P_w1a,  Cc, HIDc, totCH);
  pack_w_k<<<PB_CH, 256, 0, stream>>>(w1b,        P_w1b,  Cc, HIDc, totCH);
  pack_w_k<<<PB_HC, 256, 0, stream>>>(w2,         P_w2,   HIDc, Cc, totHC);

  for (int blk = 0; blk < NBc; ++blk) {
    const size_t cc = (size_t)Cc * Cc, ch = (size_t)Cc * HIDc, hc = (size_t)HIDc * Cc;
    const float* p_adaln = adaln_s_w   + (size_t)blk * Cc;
    const float* p_gb    = gate_b      + (size_t)blk * Cc;
    const float* p_qb    = q_b         + (size_t)blk * Cc;
    const float* p_plw   = pair_ln_w   + (size_t)blk * CPc;
    const float* p_plb   = pair_ln_b   + (size_t)blk * CPc;
    const float* p_pbw   = pair_b_w    + (size_t)blk * CPc * Hc;
    const float* p_sgb   = s_gate_b    + (size_t)blk * Cc;
    const float* p_tad   = t_adaln_s_w + (size_t)blk * Cc;
    const float* p_tgb   = t_gate_b    + (size_t)blk * Cc;
    const float* p_tsgb  = t_s_gate_b  + (size_t)blk * Cc;

    ln_k<<<Mc, 128, 0, stream>>>(a_buf, a_ln);
    scale_bf16_k<<<ELT_B, 256, 0, stream>>>(s_ln, p_adaln, s_n);
    scale_bf16_k<<<ELT_B, 256, 0, stream>>>(s_ln, p_tad, s_t);

    adaln_gemm_k<<<ADA_CC, 256, 0, stream>>>(s_n, P_gw + blk * cc, p_gb,
                                             P_sk + blk * cc, a_ln, a_n,
                                             Mc, Cc, Cc);
    gemm_k<true,  false, false, false, 1><<<GEMM64_C, 256, 0, stream>>>(
        a_n, P_qw + blk * cc, p_qb, nullptr, nullptr, qb, Mc, Cc, Cc);
    gemm_k<false, false, false, false, 1><<<GEMM64_C, 256, 0, stream>>>(
        a_n, P_kw + blk * cc, nullptr, nullptr, nullptr, kb, Mc, Cc, Cc);
    gemm_k<false, false, false, false, 2><<<GEMM64_C, 256, 0, stream>>>(
        a_n, P_vw + blk * cc, nullptr, nullptr, nullptr, vtb, Mc, Cc, Cc);
    gemm_k<false, true,  false, false, 0><<<GEMM64_C, 256, 0, stream>>>(
        a_n, P_gat + blk * cc, nullptr, nullptr, gbuf, nullptr, Mc, Cc, Cc);
    gemm_k<true,  true,  false, false, 0><<<GEMM64_C, 256, 0, stream>>>(
        s_n, P_sgw + blk * cc, p_sgb, nullptr, sg, nullptr, Mc, Cc, Cc);

    zbias_k<<<ZB_B, 256, 0, stream>>>(pair_repr, p_plw, p_plb, p_pbw, zbuf);
    attn_k<<<ATTN_B, 32, 0, stream>>>(qb, kb, vtb, zbuf, mask, obuf);

    mul_bf16_k<<<ELT_B, 256, 0, stream>>>(obuf, gbuf, ogb);
    gemm_k<false, false, true,  false, 0><<<GEMM64_C, 256, 0, stream>>>(
        ogb, P_ow + blk * cc, nullptr, sg, a_buf, nullptr, Mc, Cc, Cc);

    adaln_gemm_k<<<ADA_CC, 256, 0, stream>>>(s_t, P_tgw + blk * cc, p_tgb,
                                             P_tsk + blk * cc, a_ln, a_t,
                                             Mc, Cc, Cc);
    swiglu_gemm_k<<<SWI_CH, 256, 0, stream>>>(a_t, P_w1a + blk * ch,
                                              P_w1b + blk * ch, hidden,
                                              Mc, Cc, HIDc);
    gemm_k<true,  true,  false, false, 0><<<GEMM64_C, 256, 0, stream>>>(
        s_t, P_tsgw + blk * cc, p_tsgb, nullptr, tsg, nullptr, Mc, Cc, Cc);
    gemm_k<false, false, true,  true,  0><<<GEMM64_C, 256, 0, stream>>>(
        hidden, P_w2 + blk * hc, nullptr, tsg, a_buf, nullptr, Mc, HIDc, Cc);
  }

  copy_k<<<(out_size + 255) / 256, 256, 0, stream>>>(a_buf, (float*)d_out, out_size);
}
